// Qwen3_5MTP_29454885716646
// MI455X (gfx1250) — compile-verified, matrix operand order checked by reading, not run on hardware
//
#include <hip/hip_runtime.h>

#define T_DIM 2048
#define H_DIM 2048
#define NH 16
#define NKV 4
#define DH 128
#define NHD (NH*DH)   // 2048
#define KVD (NKV*DH)  // 512
#define E_DIM 16
#define I_DIM 1024
#define EPS_F 1e-6f
#define THETA_F 1000000.0f

typedef __bf16 bf16_t;
typedef __attribute__((ext_vector_type(16))) __bf16 bf16x16;
typedef __attribute__((ext_vector_type(8)))  __bf16 bf16x8;
typedef __attribute__((ext_vector_type(2)))  __bf16 bf16x2;
typedef __attribute__((ext_vector_type(8)))  float  floatx8;

// ---------------------------------------------------------------------------
// WMMA helpers (CDNA5 16x16x32 bf16, f32 accumulate).
// A-fragment: lane m=lane&15 owns row m; element i -> K = kb+i (i<8),
// 16+kb+(i-8) (i>=8), kb = lane<16 ? 0 : 8.  B symmetric with N across lanes.
// C/D: lane n=lane&15, vgpr r -> row r + (lane<16?0:8).
// ---------------------------------------------------------------------------
__device__ __forceinline__ floatx8 wmma_bf16(bf16x16 a, bf16x16 b, floatx8 c) {
  return __builtin_amdgcn_wmma_f32_16x16x32_bf16(false, a, false, b, (short)0, c,
                                                 false, false);
}

// Fragment from a contiguous row of 32 K-values: two 128-bit loads.
// Requires 16-byte alignment of p (kb is 0 or 8 -> 0/16B offset).
__device__ __forceinline__ bf16x16 fragA_contig(const bf16_t* p, int kb) {
  bf16x8 lo = *(const bf16x8*)(p + kb);
  bf16x8 hi = *(const bf16x8*)(p + 16 + kb);
  return __builtin_shufflevector(lo, hi, 0, 1, 2, 3, 4, 5, 6, 7,
                                 8, 9, 10, 11, 12, 13, 14, 15);
}
// B fragment with K strided by ld (fp32 weights, convert in-register).
__device__ __forceinline__ bf16x16 fragB_strided_f32(const float* p, int ld, int kb) {
  bf16x16 v;
#pragma unroll
  for (int i = 0; i < 8; i++) {
    v[i]     = (bf16_t)p[(kb + i) * ld];
    v[i + 8] = (bf16_t)p[(16 + kb + i) * ld];
  }
  return v;
}

// ---------------------------------------------------------------------------
// Kernel 1: gather embedding row, RMS-norm emb and hidden, emit bf16 concat.
// ---------------------------------------------------------------------------
__global__ __launch_bounds__(256) void build_xcat_kernel(
    const int* __restrict__ ids, const float* __restrict__ embed_w,
    const float* __restrict__ hid, const float* __restrict__ w_emb,
    const float* __restrict__ w_hid, bf16_t* __restrict__ xcat) {
  int t = blockIdx.x, tid = threadIdx.x;
  const float* erow = embed_w + (size_t)ids[t] * H_DIM;
  const float* hrow = hid + (size_t)t * H_DIM;
  __shared__ float red[256];

  float ssq = 0.f;
  for (int k = tid; k < H_DIM; k += 256) { float v = erow[k]; ssq += v * v; }
  red[tid] = ssq; __syncthreads();
  for (int s = 128; s > 0; s >>= 1) { if (tid < s) red[tid] += red[tid + s]; __syncthreads(); }
  float r = rsqrtf(red[0] * (1.0f / H_DIM) + EPS_F);
  __syncthreads();
  for (int k = tid; k < H_DIM; k += 256)
    xcat[(size_t)t * 4096 + k] = (bf16_t)(erow[k] * r * (1.f + w_emb[k]));

  ssq = 0.f;
  for (int k = tid; k < H_DIM; k += 256) { float v = hrow[k]; ssq += v * v; }
  red[tid] = ssq; __syncthreads();
  for (int s = 128; s > 0; s >>= 1) { if (tid < s) red[tid] += red[tid + s]; __syncthreads(); }
  r = rsqrtf(red[0] * (1.0f / H_DIM) + EPS_F);
  for (int k = tid; k < H_DIM; k += 256)
    xcat[(size_t)t * 4096 + H_DIM + k] = (bf16_t)(hrow[k] * r * (1.f + w_hid[k]));
}

// ---------------------------------------------------------------------------
// Generic WMMA GEMM: C[M,N] (f32) = A_bf16[M,K] @ B_f32[K,N] (+ addend).
// Block tile 64x64x64, 8 waves (2x4), each wave owns two 16x16 M-subtiles.
// A staged row-major, B staged TRANSPOSED (n-major) so both operands use
// vectorized b128 fragment loads.  Staging loads are gathered into registers
// FIRST (batched clause, many outstanding LOADcnt) and only then stored to
// LDS; B reads are float4 (b128) and writes are packed bf16x2 (b32).
// Rows padded to 80 elems (160B) for alignment + bank spread.
// ---------------------------------------------------------------------------
#define BKP 80
__global__ __launch_bounds__(256) void gemm_bf16_kernel(
    const bf16_t* __restrict__ A, const float* __restrict__ B,
    const float* __restrict__ addend, float* __restrict__ C,
    int M, int N, int Kd) {
  __shared__ bf16_t Ash[64 * BKP];
  __shared__ bf16_t Bsh[64 * BKP];
  int tid = threadIdx.x;
  int wave = tid >> 5, lane = tid & 31;
  int wm = wave >> 2, wn = wave & 3;
  int m = lane & 15;
  int kb = (lane < 16) ? 0 : 8;
  int moff = (lane < 16) ? 0 : 8;
  int row0 = blockIdx.y * 64, col0 = blockIdx.x * 64;
  (void)M;

  floatx8 acc[2] = {{}, {}};
  for (int k0 = 0; k0 < Kd; k0 += 64) {
    // ---- gather phase: issue ALL global loads into registers first --------
    bf16x8 areg[2];
#pragma unroll
    for (int j = 0; j < 2; j++) {            // A: 64x64 bf16, 2 b128 per thread
      int vidx = tid + j * 256;              // 512 vectors of 8 bf16
      int r = vidx >> 3;
      int kv = (vidx & 7) * 8;
      areg[j] = *(const bf16x8*)&A[(size_t)(row0 + r) * Kd + k0 + kv];
    }
    float4 breg[4];
#pragma unroll
    for (int j = 0; j < 2; j++) {            // B: 64x64 f32, 4 float4 per thread
      int idx = tid + j * 256;               // 512 (k-pair, n4) items
      int p = idx >> 4;                      // k-row pair 0..31
      int n4 = (idx & 15) * 4;               // 4 consecutive columns
      breg[2 * j + 0] = *(const float4*)&B[(size_t)(k0 + 2 * p)     * N + col0 + n4];
      breg[2 * j + 1] = *(const float4*)&B[(size_t)(k0 + 2 * p + 1) * N + col0 + n4];
    }
    if (k0 + 64 < Kd)                        // prefetch next B tile
      __builtin_prefetch(&B[(size_t)(k0 + 64) * N + col0 + (tid & 63)], 0, 1);
    // ---- store phase: registers -> LDS ------------------------------------
#pragma unroll
    for (int j = 0; j < 2; j++) {
      int vidx = tid + j * 256;
      int r = vidx >> 3;
      int kv = (vidx & 7) * 8;
      *(bf16x8*)&Ash[r * BKP + kv] = areg[j];
    }
#pragma unroll
    for (int j = 0; j < 2; j++) {
      int idx = tid + j * 256;
      int p = idx >> 4;
      int n4 = (idx & 15) * 4;
      const float* r0 = (const float*)&breg[2 * j + 0];
      const float* r1 = (const float*)&breg[2 * j + 1];
#pragma unroll
      for (int i = 0; i < 4; i++) {          // transposed, packed pair along k
        bf16x2 w2;
        w2[0] = (bf16_t)r0[i];
        w2[1] = (bf16_t)r1[i];
        *(bf16x2*)&Bsh[(n4 + i) * BKP + 2 * p] = w2;
      }
    }
    __syncthreads();
#pragma unroll
    for (int kk = 0; kk < 64; kk += 32) {
      bf16x16 bf = fragA_contig(&Bsh[(wn * 16 + m) * BKP + kk], kb);
#pragma unroll
      for (int s = 0; s < 2; s++) {
        bf16x16 af = fragA_contig(&Ash[(wm * 16 + s * 32 + m) * BKP + kk], kb);
        acc[s] = wmma_bf16(af, bf, acc[s]);
      }
    }
    __syncthreads();
  }
#pragma unroll
  for (int s = 0; s < 2; s++) {
#pragma unroll
    for (int r = 0; r < 8; r++) {
      int row = row0 + wm * 16 + s * 32 + r + moff;
      int col = col0 + wn * 16 + m;
      float v = acc[s][r];
      if (addend) v += addend[(size_t)row * N + col];
      C[(size_t)row * N + col] = v;
    }
  }
}

// ---------------------------------------------------------------------------
// RMS-norm (f32 in -> bf16 out), one block per token.
// ---------------------------------------------------------------------------
__global__ __launch_bounds__(256) void rmsnorm_bf16_kernel(
    const float* __restrict__ x, const float* __restrict__ w, bf16_t* __restrict__ out) {
  int t = blockIdx.x, tid = threadIdx.x;
  const float* row = x + (size_t)t * H_DIM;
  __shared__ float red[256];
  float ssq = 0.f;
  for (int k = tid; k < H_DIM; k += 256) { float v = row[k]; ssq += v * v; }
  red[tid] = ssq; __syncthreads();
  for (int s = 128; s > 0; s >>= 1) { if (tid < s) red[tid] += red[tid + s]; __syncthreads(); }
  float r = rsqrtf(red[0] * (1.0f / H_DIM) + EPS_F);
  for (int k = tid; k < H_DIM; k += 256)
    out[(size_t)t * H_DIM + k] = (bf16_t)(row[k] * r * (1.f + w[k]));
}

// ---------------------------------------------------------------------------
// Per-(token, head): q/k RMS-norm over DH + RoPE, v transposed passthrough.
// blockIdx.y: [0,16) q heads, [16,20) k heads, [20,24) v heads.
// v is written TRANSPOSED as vT[(kv*DH + d) * T + t] so attention's P@V
// B-fragments are contiguous over keys (vector b128 loads).
// ---------------------------------------------------------------------------
__global__ __launch_bounds__(128) void qkv_norm_rope_kernel(
    const float* __restrict__ qbuf, const float* __restrict__ kbuf,
    const float* __restrict__ vbuf, const float* __restrict__ qw,
    const float* __restrict__ kw, const int* __restrict__ pos,
    bf16_t* __restrict__ qr, bf16_t* __restrict__ kr, bf16_t* __restrict__ vT) {
  int t = blockIdx.x, hh = blockIdx.y, tid = threadIdx.x;
  if (hh >= NH + NKV) {                       // v: transpose-scatter, no norm
    int kv = hh - NH - NKV;
    float x = vbuf[(size_t)t * KVD + kv * DH + tid];
    vT[((size_t)kv * DH + tid) * T_DIM + t] = (bf16_t)x;
    return;
  }
  const float* src; bf16_t* dst; const float* w;
  if (hh < NH) {
    src = qbuf + (size_t)t * NHD + hh * DH; dst = qr + (size_t)t * NHD + hh * DH;
    w = qw;
  } else {
    int k = hh - NH;
    src = kbuf + (size_t)t * KVD + k * DH; dst = kr + (size_t)t * KVD + k * DH;
    w = kw;
  }
  float x = src[tid];
  __shared__ float red[128];
  __shared__ float xs[128];
  red[tid] = x * x; __syncthreads();
  for (int s = 64; s > 0; s >>= 1) { if (tid < s) red[tid] += red[tid + s]; __syncthreads(); }
  float r = rsqrtf(red[0] * (1.0f / DH) + EPS_F);
  xs[tid] = x * r * (1.f + w[tid]);
  __syncthreads();
  if (tid < DH / 2) {
    float p = (float)pos[t];
    float inv = __powf(THETA_F, -((float)(2 * tid)) * (1.0f / DH));
    float sn, cs; __sincosf(p * inv, &sn, &cs);
    float x1 = xs[tid], x2 = xs[tid + DH / 2];
    dst[tid]          = (bf16_t)(x1 * cs - x2 * sn);
    dst[tid + DH / 2] = (bf16_t)(x2 * cs + x1 * sn);
  }
}

// ---------------------------------------------------------------------------
// Flash attention, causal (positions == arange). One wave = 16 queries of one
// head; 32-key steps; scores + P@V via WMMA; S->P transpose via per-wave LDS.
// P rows padded to 40 elems (80B, 16B-aligned) for b128 fragment loads.
// ---------------------------------------------------------------------------
#define PPAD 40
__global__ __launch_bounds__(256) void attn_kernel(
    const bf16_t* __restrict__ qr, const bf16_t* __restrict__ kr,
    const bf16_t* __restrict__ vT, bf16_t* __restrict__ attn_out) {
  const float scale = 0.088388347648318447f;  // 1/sqrt(128)
  int h = blockIdx.x;
  int kh = h / (NH / NKV);
  int wave = threadIdx.x >> 5, lane = threadIdx.x & 31;
  int m = lane & 15;
  int kb = (lane < 16) ? 0 : 8;
  int moff = (lane < 16) ? 0 : 8;
  int q0 = blockIdx.y * 128 + wave * 16;

  __shared__ float  Ssh[8][16 * 32];
  __shared__ bf16_t Psh[8][16 * PPAD];
  __shared__ float  Csh[8][16];
  float*  S  = Ssh[wave];
  bf16_t* P  = Psh[wave];
  float*  SC = Csh[wave];

  const bf16_t* qrow = qr + (size_t)(q0 + m) * NHD + h * DH;
  bf16x16 qf[4];
#pragma unroll
  for (int c = 0; c < 4; c++) qf[c] = fragA_contig(qrow + c * 32, kb);

  floatx8 zf = {};
  floatx8 o[8];
#pragma unroll
  for (int nt = 0; nt < 8; nt++) o[nt] = zf;

  float rm = -1e30f, rl = 0.f;
  int qmax = q0 + 15;
  for (int kt = 0; kt <= qmax; kt += 32) {
    floatx8 s0 = {}, s1 = {};
    const bf16_t* k0p = kr + (size_t)(kt + m) * KVD + kh * DH;
    const bf16_t* k1p = kr + (size_t)(kt + 16 + m) * KVD + kh * DH;
#pragma unroll
    for (int c = 0; c < 4; c++) {
      s0 = wmma_bf16(qf[c], fragA_contig(k0p + c * 32, kb), s0);
      s1 = wmma_bf16(qf[c], fragA_contig(k1p + c * 32, kb), s1);
    }
#pragma unroll
    for (int r = 0; r < 8; r++) {
      S[(r + moff) * 32 + m]      = s0[r];
      S[(r + moff) * 32 + 16 + m] = s1[r];
    }
    __builtin_amdgcn_wave_barrier();
    asm volatile("s_wait_dscnt 0" ::: "memory");

    if (lane < 16) {                    // lane m owns query row q0+m
      int qi = q0 + m;
      float mx = rm;
#pragma unroll
      for (int j = 0; j < 32; j++) {
        float s = S[m * 32 + j] * scale;
        if (kt + j > qi) s = -1e30f;    // causal mask
        S[m * 32 + j] = s;
        mx = fmaxf(mx, s);
      }
      float fac = __expf(rm - mx);
      float sum = 0.f;
#pragma unroll
      for (int j = 0; j < 32; j++) {
        float p = __expf(S[m * 32 + j] - mx);
        sum += p;
        P[m * PPAD + j] = (bf16_t)p;
      }
      rl = rl * fac + sum;
      rm = mx;
      SC[m] = fac;
    }
    __builtin_amdgcn_wave_barrier();
    asm volatile("s_wait_dscnt 0" ::: "memory");

    float fv[8];
#pragma unroll
    for (int r = 0; r < 8; r++) fv[r] = SC[r + moff];
#pragma unroll
    for (int nt = 0; nt < 8; nt++) {
#pragma unroll
      for (int r = 0; r < 8; r++) o[nt][r] *= fv[r];
    }
    bf16x16 pf = fragA_contig(P + m * PPAD, kb);
#pragma unroll
    for (int nt = 0; nt < 8; nt++) {
      // vT[(kh*DH + d) * T + key]: contiguous over keys -> b128 loads
      const bf16_t* vp = vT + ((size_t)kh * DH + nt * 16 + m) * T_DIM + kt;
      o[nt] = wmma_bf16(pf, fragA_contig(vp, kb), o[nt]);
    }
    __builtin_amdgcn_wave_barrier();
  }
  if (lane < 16) SC[m] = 1.f / rl;
  __builtin_amdgcn_wave_barrier();
  asm volatile("s_wait_dscnt 0" ::: "memory");
  float lv[8];
#pragma unroll
  for (int r = 0; r < 8; r++) lv[r] = SC[r + moff];
#pragma unroll
  for (int nt = 0; nt < 8; nt++) {
#pragma unroll
    for (int r = 0; r < 8; r++) {
      attn_out[(size_t)(q0 + r + moff) * NHD + h * DH + nt * 16 + m] =
          (bf16_t)(o[nt][r] * lv[r]);
    }
  }
}

// ---------------------------------------------------------------------------
// Router: logits, softmax top-2 (== softmax over top-2 logits), scatter
// (token,slot) into per-expert lists.
// ---------------------------------------------------------------------------
__global__ __launch_bounds__(256) void router_kernel(
    const bf16_t* __restrict__ xn2, const float* __restrict__ router_w,
    float* __restrict__ topw, int* __restrict__ ex_idx, int* __restrict__ cnt) {
  int t = blockIdx.x, tid = threadIdx.x;
  float acc[E_DIM];
#pragma unroll
  for (int e = 0; e < E_DIM; e++) acc[e] = 0.f;
  for (int k = tid; k < H_DIM; k += 256) {
    float x = (float)xn2[(size_t)t * H_DIM + k];
#pragma unroll
    for (int e = 0; e < E_DIM; e++) acc[e] += x * router_w[(size_t)k * E_DIM + e];
  }
  __shared__ float red[256];
  __shared__ float logits[E_DIM];
  for (int e = 0; e < E_DIM; e++) {
    red[tid] = acc[e]; __syncthreads();
    for (int s = 128; s > 0; s >>= 1) { if (tid < s) red[tid] += red[tid + s]; __syncthreads(); }
    if (tid == 0) logits[e] = red[0];
    __syncthreads();
  }
  if (tid == 0) {
    int i0 = 0;
    for (int e = 1; e < E_DIM; e++) if (logits[e] > logits[i0]) i0 = e;
    int i1 = (i0 == 0) ? 1 : 0;
    for (int e = 0; e < E_DIM; e++)
      if (e != i0 && logits[e] > logits[i1]) i1 = e;
    float e1 = __expf(logits[i1] - logits[i0]);
    float w0 = 1.f / (1.f + e1), w1 = e1 / (1.f + e1);
    topw[t * 2 + 0] = w0;
    topw[t * 2 + 1] = w1;
    int p0 = atomicAdd(&cnt[i0], 1); ex_idx[i0 * T_DIM + p0] = t * 2 + 0;
    int p1 = atomicAdd(&cnt[i1], 1); ex_idx[i1 * T_DIM + p1] = t * 2 + 1;
  }
}

// ---------------------------------------------------------------------------
// MoE gate/up: per (expert, 16-token tile, 64 cols of I). Gathered A in LDS
// (rows padded to 40, staged as one b128 per thread), weights streamed from
// global, silu(g)*u -> a_buf row (t*2+slot).
// ---------------------------------------------------------------------------
#define APAD 40
__global__ __launch_bounds__(128) void moe_gateup_kernel(
    const bf16_t* __restrict__ xn2, const float* __restrict__ w_gate,
    const float* __restrict__ w_up, const int* __restrict__ ex_idx,
    const int* __restrict__ cnt, bf16_t* __restrict__ a_buf) {
  int e = blockIdx.z;
  int n = cnt[e];
  int r0 = blockIdx.y * 16;
  if (r0 >= n) return;
  int wave = threadIdx.x >> 5, lane = threadIdx.x & 31;
  int m = lane & 15;
  int kb = (lane < 16) ? 0 : 8;
  int moff = (lane < 16) ? 0 : 8;
  int col0 = blockIdx.x * 64 + wave * 16;

  __shared__ int rtok[16];
  __shared__ bf16_t Atile[16 * APAD];
  if (threadIdx.x < 16) {
    int rr = r0 + threadIdx.x;
    rtok[threadIdx.x] = (rr < n) ? ex_idx[e * T_DIM + rr] : -1;
  }
  __syncthreads();
  int sr = threadIdx.x >> 2;            // staging: row 0..15 (threads 0..63)
  int sv = (threadIdx.x & 3) * 8;       // 4 b128 vectors per 32-elem row

  floatx8 g = {}, u = {};
  const float* gbase = w_gate + (size_t)e * H_DIM * I_DIM + col0 + m;
  const float* ubase = w_up   + (size_t)e * H_DIM * I_DIM + col0 + m;
  for (int k0 = 0; k0 < H_DIM; k0 += 32) {
    if (threadIdx.x < 64) {             // one b128 load + one b128 store
      int ri = rtok[sr];
      int tok = (ri >= 0) ? (ri >> 1) : 0;
      *(bf16x8*)&Atile[sr * APAD + sv] =
          *(const bf16x8*)&xn2[(size_t)tok * H_DIM + k0 + sv];
    }
    __syncthreads();
    bf16x16 af = fragA_contig(&Atile[m * APAD], kb);
    bf16x16 bg = fragB_strided_f32(gbase + (size_t)k0 * I_DIM, I_DIM, kb);
    bf16x16 bu = fragB_strided_f32(ubase + (size_t)k0 * I_DIM, I_DIM, kb);
    g = wmma_bf16(af, bg, g);
    u = wmma_bf16(af, bu, u);
    __syncthreads();
  }
#pragma unroll
  for (int r = 0; r < 8; r++) {
    int rr = r + moff;
    int ri = rtok[rr];
    if (ri >= 0) {
      float gv = g[r];
      float a = (gv / (1.f + __expf(-gv))) * u[r];  // silu(g)*u
      a_buf[(size_t)ri * I_DIM + col0 + m] = (bf16_t)a;
    }
  }
}

// ---------------------------------------------------------------------------
// MoE down: a_buf rows @ w_down[e], gate-scaled, atomic-accumulate into xout
// (xout pre-initialized with the attention residual).
// ---------------------------------------------------------------------------
__global__ __launch_bounds__(128) void moe_down_kernel(
    const bf16_t* __restrict__ a_buf, const float* __restrict__ w_down,
    const float* __restrict__ topw, const int* __restrict__ ex_idx,
    const int* __restrict__ cnt, float* __restrict__ xout) {
  int e = blockIdx.z;
  int n = cnt[e];
  int r0 = blockIdx.y * 16;
  if (r0 >= n) return;
  int wave = threadIdx.x >> 5, lane = threadIdx.x & 31;
  int m = lane & 15;
  int kb = (lane < 16) ? 0 : 8;
  int moff = (lane < 16) ? 0 : 8;
  int col0 = blockIdx.x * 64 + wave * 16;

  __shared__ int rtok[16];
  __shared__ float gts[16];
  __shared__ bf16_t Atile[16 * APAD];
  if (threadIdx.x < 16) {
    int rr = r0 + threadIdx.x;
    int ri = (rr < n) ? ex_idx[e * T_DIM + rr] : -1;
    rtok[threadIdx.x] = ri;
    gts[threadIdx.x] = (ri >= 0) ? topw[ri] : 0.f;
  }
  __syncthreads();
  int sr = threadIdx.x >> 2;
  int sv = (threadIdx.x & 3) * 8;

  floatx8 acc = {};
  const float* dbase = w_down + (size_t)e * I_DIM * H_DIM + col0 + m;
  for (int k0 = 0; k0 < I_DIM; k0 += 32) {
    if (threadIdx.x < 64) {
      int ri = rtok[sr];
      int arow = (ri >= 0) ? ri : 0;
      *(bf16x8*)&Atile[sr * APAD + sv] =
          *(const bf16x8*)&a_buf[(size_t)arow * I_DIM + k0 + sv];
    }
    __syncthreads();
    bf16x16 af = fragA_contig(&Atile[m * APAD], kb);
    bf16x16 bd = fragB_strided_f32(dbase + (size_t)k0 * H_DIM, H_DIM, kb);
    acc = wmma_bf16(af, bd, acc);
    __syncthreads();
  }
#pragma unroll
  for (int r = 0; r < 8; r++) {
    int rr = r + moff;
    int ri = rtok[rr];
    if (ri >= 0)
      atomicAdd(&xout[(size_t)(ri >> 1) * H_DIM + col0 + m], acc[r] * gts[rr]);
  }
}

// ---------------------------------------------------------------------------
// Small utility kernels.
// ---------------------------------------------------------------------------
__global__ void zero_cnt_kernel(int* c) { if (threadIdx.x < E_DIM) c[threadIdx.x] = 0; }

__global__ __launch_bounds__(256) void copyf4_kernel(const float4* __restrict__ src,
                                                     float4* __restrict__ dst, int n) {
  int i = blockIdx.x * 256 + threadIdx.x;
  if (i < n) dst[i] = src[i];
}

__global__ __launch_bounds__(256) void final_rms_kernel(
    const float* __restrict__ x, const float* __restrict__ w, float* __restrict__ out) {
  int t = blockIdx.x, tid = threadIdx.x;
  const float* row = x + (size_t)t * H_DIM;
  __shared__ float red[256];
  float ssq = 0.f;
  for (int k = tid; k < H_DIM; k += 256) { float v = row[k]; ssq += v * v; }
  red[tid] = ssq; __syncthreads();
  for (int s = 128; s > 0; s >>= 1) { if (tid < s) red[tid] += red[tid + s]; __syncthreads(); }
  float r = rsqrtf(red[0] * (1.0f / H_DIM) + EPS_F);
  for (int k = tid; k < H_DIM; k += 256)
    out[(size_t)t * H_DIM + k] = row[k] * r * (1.f + w[k]);
}

// ---------------------------------------------------------------------------
extern "C" void kernel_launch(void* const* d_in, const int* in_sizes, int n_in,
                              void* d_out, int out_size, void* d_ws, size_t ws_size,
                              hipStream_t stream) {
  (void)in_sizes; (void)n_in; (void)out_size; (void)ws_size;
  const int*   input_ids    = (const int*)  d_in[0];
  const int*   positions    = (const int*)  d_in[1];
  const float* hidden       = (const float*)d_in[2];
  const float* embed_w      = (const float*)d_in[4];
  const float* fc_w         = (const float*)d_in[5];
  const float* pre_fc_emb_w = (const float*)d_in[6];
  const float* pre_fc_hid_w = (const float*)d_in[7];
  const float* in_ln_w      = (const float*)d_in[8];
  const float* post_ln_w    = (const float*)d_in[9];
  const float* final_norm_w = (const float*)d_in[10];
  const float* wq           = (const float*)d_in[11];
  const float* wk           = (const float*)d_in[12];
  const float* wv           = (const float*)d_in[13];
  const float* wo           = (const float*)d_in[14];
  const float* q_norm_w     = (const float*)d_in[15];
  const float* k_norm_w     = (const float*)d_in[16];
  const float* router_w     = (const float*)d_in[17];
  const float* w_gate       = (const float*)d_in[18];
  const float* w_up         = (const float*)d_in[19];
  const float* w_down       = (const float*)d_in[20];
  float* out = (float*)d_out;

  char* base = (char*)d_ws;
  size_t off = 0;
  auto alloc = [&](size_t bytes) -> void* {
    void* p = base + off;
    off += (bytes + 255) & ~(size_t)255;
    return p;
  };
  bf16_t* xcat  = (bf16_t*)alloc((size_t)T_DIM * 4096 * sizeof(bf16_t));
  float*  res1  = (float*) alloc((size_t)T_DIM * H_DIM * sizeof(float));
  bf16_t* xn    = (bf16_t*)alloc((size_t)T_DIM * H_DIM * sizeof(bf16_t));
  float*  qbuf  = (float*) alloc((size_t)T_DIM * NHD * sizeof(float));
  float*  kbuf  = (float*) alloc((size_t)T_DIM * KVD * sizeof(float));
  float*  vbuf  = (float*) alloc((size_t)T_DIM * KVD * sizeof(float));
  bf16_t* qrb   = (bf16_t*)alloc((size_t)T_DIM * NHD * sizeof(bf16_t));
  bf16_t* krb   = (bf16_t*)alloc((size_t)T_DIM * KVD * sizeof(bf16_t));
  bf16_t* vTb   = (bf16_t*)alloc((size_t)KVD * T_DIM * sizeof(bf16_t));
  bf16_t* attn  = (bf16_t*)alloc((size_t)T_DIM * NHD * sizeof(bf16_t));
  float*  res2  = (float*) alloc((size_t)T_DIM * H_DIM * sizeof(float));
  bf16_t* xn2   = (bf16_t*)alloc((size_t)T_DIM * H_DIM * sizeof(bf16_t));
  float*  topw  = (float*) alloc((size_t)T_DIM * 2 * sizeof(float));
  int*    exidx = (int*)   alloc((size_t)E_DIM * T_DIM * sizeof(int));
  int*    cnt   = (int*)   alloc(64 * sizeof(int));
  bf16_t* a_buf = (bf16_t*)alloc((size_t)T_DIM * 2 * I_DIM * sizeof(bf16_t));
  float*  xout  = (float*) alloc((size_t)T_DIM * H_DIM * sizeof(float));

  // 1) concat(rms(emb), rms(hid)) -> bf16
  build_xcat_kernel<<<T_DIM, 256, 0, stream>>>(input_ids, embed_w, hidden,
                                               pre_fc_emb_w, pre_fc_hid_w, xcat);
  // 2) fc: (T,4096) @ (4096,2048) -> residual1
  gemm_bf16_kernel<<<dim3(H_DIM / 64, T_DIM / 64), 256, 0, stream>>>(
      xcat, fc_w, nullptr, res1, T_DIM, H_DIM, 2 * H_DIM);
  // 3) input layernorm
  rmsnorm_bf16_kernel<<<T_DIM, 256, 0, stream>>>(res1, in_ln_w, xn);
  // 4) q/k/v projections
  gemm_bf16_kernel<<<dim3(NHD / 64, T_DIM / 64), 256, 0, stream>>>(
      xn, wq, nullptr, qbuf, T_DIM, NHD, H_DIM);
  gemm_bf16_kernel<<<dim3(KVD / 64, T_DIM / 64), 256, 0, stream>>>(
      xn, wk, nullptr, kbuf, T_DIM, KVD, H_DIM);
  gemm_bf16_kernel<<<dim3(KVD / 64, T_DIM / 64), 256, 0, stream>>>(
      xn, wv, nullptr, vbuf, T_DIM, KVD, H_DIM);
  // 5) per-head RMS norm + RoPE (+ v transpose)
  qkv_norm_rope_kernel<<<dim3(T_DIM, NH + 2 * NKV), 128, 0, stream>>>(
      qbuf, kbuf, vbuf, q_norm_w, k_norm_w, positions, qrb, krb, vTb);
  // 6) causal flash attention
  attn_kernel<<<dim3(NH, T_DIM / 128), 256, 0, stream>>>(qrb, krb, vTb, attn);
  // 7) output projection + residual
  gemm_bf16_kernel<<<dim3(H_DIM / 64, T_DIM / 64), 256, 0, stream>>>(
      attn, wo, res1, res2, T_DIM, H_DIM, NHD);
  // 8) post layernorm
  rmsnorm_bf16_kernel<<<T_DIM, 256, 0, stream>>>(res2, post_ln_w, xn2);
  // 9) routing
  zero_cnt_kernel<<<1, 64, 0, stream>>>(cnt);
  router_kernel<<<T_DIM, 256, 0, stream>>>(xn2, router_w, topw, exidx, cnt);
  // 10) xout = residual2 (MoE output accumulates on top)
  copyf4_kernel<<<(T_DIM * H_DIM / 4 + 255) / 256, 256, 0, stream>>>(
      (const float4*)res2, (float4*)xout, T_DIM * H_DIM / 4);
  // 11) MoE gate/up and down (gathered per-expert GEMMs, 8x fewer FLOPs)
  moe_gateup_kernel<<<dim3(I_DIM / 64, T_DIM / 16, E_DIM), 128, 0, stream>>>(
      xn2, w_gate, w_up, exidx, cnt, a_buf);
  moe_down_kernel<<<dim3(H_DIM / 64, T_DIM / 16, E_DIM), 128, 0, stream>>>(
      a_buf, w_down, topw, exidx, cnt, xout);
  // 12) final norm -> output
  final_rms_kernel<<<T_DIM, 256, 0, stream>>>(xout, final_norm_w, out);
}
